// GAT_DGL_Custom_55594056680299
// MI455X (gfx1250) — matrix operand order repaired and checked by reading, not compiled, not run on hardware
//
#include <hip/hip_runtime.h>
#include <hip/hip_bf16.h>
#include <math.h>

// ---------------------------------------------------------------------------
// GAT (2-layer DGL GATConv) for MI455X / gfx1250.
//  - Dense GEMMs via v_wmma_f32_16x16x32_f16 (wave32, 16x16 C tile per wave).
//  - Edge softmax/aggregation via L2-resident gathers + float atomics
//    (whole working set ~115MB < 192MB L2).
// ---------------------------------------------------------------------------

typedef _Float16 half_t;
typedef __attribute__((ext_vector_type(16))) _Float16 v16h;
typedef __attribute__((ext_vector_type(8)))  float    v8f;

#define NN 50000
#define NE 800000
#define FIN 256          // IN_CH
#define HD1 256          // H1*D1
#define H1C 4
#define HD2 64           // H2*D2
#define H2C 1
#define NEG_SLOPE 0.2f

// ---------------- float atomic max (int-max / uint-min trick) --------------
// emax buffer is initialized to 0xFFFFFFFF bytes (memsetAsync), which loses
// to any update on both paths.
__device__ __forceinline__ void atomicMaxF(float* addr, float val) {
  if (val >= 0.0f) {
    atomicMax((int*)addr, __float_as_int(val));
  } else {
    atomicMin((unsigned int*)addr, __float_as_uint(val));
  }
}

// ---------------------------------------------------------------------------
// WMMA GEMM: C[M,N] = A[M,K] * B[K,N], all f32 row-major, f16 multiplicands,
// f32 accumulation. Block = 128 threads = 4 waves; wave w computes the 16x16
// tile at (blockIdx.y*16, (blockIdx.x*4+w)*16). M,K,N exact multiples of
// 16/32/64 -> no predication, EXEC all ones (WMMA requirement).
//
// Fragment layouts used here (contiguous K in A and B, not the ISA table order):
//  A 16x32 f16 : lane L (row = L&15), K-halves: lanes0-15 hold K 0-15,
//                lanes16-31 hold K 16-31, two K per VGPR.
//  B 32x16 f16 : col = lane&15; lanes0-15 hold K 0-15, lanes16-31 K 16-31.
//  C/D 16x16 f32: col = lane&15; VGPR v holds row v (+8 for lanes16-31).
// ---------------------------------------------------------------------------
__global__ __launch_bounds__(128) void k_wmma_gemm(
    const float* __restrict__ A, const float* __restrict__ B,
    float* __restrict__ C, int K, int N) {
  const int lane = threadIdx.x & 31;
  const int wave = threadIdx.x >> 5;
  const int row  = lane & 15;           // A row / B,C col within tile
  const int hi   = lane >> 4;           // lane group 0/1
  const int m0   = blockIdx.y << 4;
  const int n0   = ((blockIdx.x << 2) + wave) << 4;
  const int akb  = hi ? 16 : 0;
  const int bkb  = hi ? 16 : 0;

  v8f acc = {};
  for (int k0 = 0; k0 < K; k0 += 32) {
    v16h a, b;
    const float* ap = A + (size_t)(m0 + row) * K + k0;
#pragma unroll
    for (int v = 0; v < 8; ++v) {
      const int k = akb + 2 * v;
      const float2 f = *(const float2*)(ap + k);
      a[2 * v]     = (half_t)f.x;
      a[2 * v + 1] = (half_t)f.y;
    }
    const float* bp = B + (size_t)(k0 + bkb) * N + n0 + row;
#pragma unroll
    for (int v = 0; v < 8; ++v) {
      b[2 * v]     = (half_t)bp[(2 * v) * N];
      b[2 * v + 1] = (half_t)bp[(2 * v + 1) * N];
    }
    acc = __builtin_amdgcn_wmma_f32_16x16x32_f16(
        false, a, false, b, (short)0, acc, false, false);
  }
  float* cp = C + n0 + row;
#pragma unroll
  for (int v = 0; v < 8; ++v)
    cp[(size_t)(m0 + v + hi * 8) * N] = acc[v];
}

// Dual-B GEMM for layer 2: ft2 = A@B1 and res2 = A@B2 share the A fragments.
__global__ __launch_bounds__(128) void k_wmma_gemm_dual(
    const float* __restrict__ A, const float* __restrict__ B1,
    const float* __restrict__ B2, float* __restrict__ C1,
    float* __restrict__ C2, int K, int N) {
  const int lane = threadIdx.x & 31;
  const int wave = threadIdx.x >> 5;
  const int row  = lane & 15;
  const int hi   = lane >> 4;
  const int m0   = blockIdx.y << 4;
  const int n0   = wave << 4;           // N=64 -> 4 waves cover all columns
  const int akb  = hi ? 16 : 0;
  const int bkb  = hi ? 16 : 0;

  v8f acc1 = {}, acc2 = {};
  for (int k0 = 0; k0 < K; k0 += 32) {
    v16h a, b1, b2;
    const float* ap = A + (size_t)(m0 + row) * K + k0;
#pragma unroll
    for (int v = 0; v < 8; ++v) {
      const int k = akb + 2 * v;
      const float2 f = *(const float2*)(ap + k);
      a[2 * v]     = (half_t)f.x;
      a[2 * v + 1] = (half_t)f.y;
    }
    const float* bp1 = B1 + (size_t)(k0 + bkb) * N + n0 + row;
    const float* bp2 = B2 + (size_t)(k0 + bkb) * N + n0 + row;
#pragma unroll
    for (int v = 0; v < 8; ++v) {
      b1[2 * v]     = (half_t)bp1[(2 * v) * N];
      b1[2 * v + 1] = (half_t)bp1[(2 * v + 1) * N];
      b2[2 * v]     = (half_t)bp2[(2 * v) * N];
      b2[2 * v + 1] = (half_t)bp2[(2 * v + 1) * N];
    }
    acc1 = __builtin_amdgcn_wmma_f32_16x16x32_f16(
        false, a, false, b1, (short)0, acc1, false, false);
    acc2 = __builtin_amdgcn_wmma_f32_16x16x32_f16(
        false, a, false, b2, (short)0, acc2, false, false);
  }
  float* c1 = C1 + n0 + row;
  float* c2 = C2 + n0 + row;
#pragma unroll
  for (int v = 0; v < 8; ++v) {
    c1[(size_t)(m0 + v + hi * 8) * N] = acc1[v];
    c2[(size_t)(m0 + v + hi * 8) * N] = acc2[v];
  }
}

// ---------------- per-node attention projections el/er ---------------------
// thread = (node, head); D = 64 fixed; 16 float4 MACs per side.
__global__ void k_el_er(const float* __restrict__ ft,
                        const float* __restrict__ al,
                        const float* __restrict__ ar,
                        float* __restrict__ el, float* __restrict__ er,
                        int H) {
  const int i = blockIdx.x * blockDim.x + threadIdx.x;
  if (i >= NN * H) return;
  const int n = i / H, h = i - n * H;
  const float4* fp = (const float4*)(ft + (size_t)n * (H * 64) + h * 64);
  const float4* lp = (const float4*)(al + h * 64);
  const float4* rp = (const float4*)(ar + h * 64);
  float sl = 0.f, sr = 0.f;
#pragma unroll
  for (int q = 0; q < 16; ++q) {
    const float4 f = fp[q], l = lp[q], r = rp[q];
    sl += f.x * l.x + f.y * l.y + f.z * l.z + f.w * l.w;
    sr += f.x * r.x + f.y * r.y + f.z * r.z + f.w * r.w;
  }
  el[i] = sl;
  er[i] = sr;
}

// ---------------- edge logits + segment max --------------------------------
__global__ void k_edge_logits_max(const int* __restrict__ src,
                                  const int* __restrict__ dst,
                                  const float* __restrict__ el,
                                  const float* __restrict__ er,
                                  float* __restrict__ ebuf,
                                  float* __restrict__ emax, int H) {
  const int i = blockIdx.x * blockDim.x + threadIdx.x;
  if (i >= NE * H) return;
  const int e = i / H, h = i - e * H;
  float x = el[src[e] * H + h] + er[dst[e] * H + h];
  x = (x >= 0.f) ? x : NEG_SLOPE * x;
  ebuf[i] = x;
  atomicMaxF(emax + dst[e] * H + h, x);
}

// ---------------- exp(e - max) + segment sum -------------------------------
__global__ void k_edge_exp_sum(const int* __restrict__ dst,
                               float* __restrict__ ebuf,
                               const float* __restrict__ emax,
                               float* __restrict__ denom, int H) {
  const int i = blockIdx.x * blockDim.x + threadIdx.x;
  if (i >= NE * H) return;
  const int e = i / H, h = i - e * H;
  const float ee = __expf(ebuf[i] - emax[dst[e] * H + h]);
  ebuf[i] = ee;
  atomicAdd(denom + dst[e] * H + h, ee);
}

// ---------------- normalized message aggregation ---------------------------
// thread = (edge, 4-feature chunk); D = 64 -> head = chunk>>4.
__global__ void k_edge_aggregate(const int* __restrict__ src,
                                 const int* __restrict__ dst,
                                 const float* __restrict__ ft,
                                 const float* __restrict__ ebuf,
                                 const float* __restrict__ denom,
                                 float* __restrict__ rst, int H, int HD) {
  const int chunks = HD >> 2;
  const long long i = (long long)blockIdx.x * blockDim.x + threadIdx.x;
  if (i >= (long long)NE * chunks) return;
  const int e = (int)(i / chunks);
  const int c = (int)(i - (long long)e * chunks);
  const int h = c >> 4;                       // (c*4)/64
  const int s = src[e], d = dst[e];
  const float a = ebuf[e * H + h] / denom[d * H + h];
  const float4 f = *(const float4*)(ft + (size_t)s * HD + (c << 2));
  float* o = rst + (size_t)d * HD + (c << 2);
  atomicAdd(o + 0, f.x * a);
  atomicAdd(o + 1, f.y * a);
  atomicAdd(o + 2, f.z * a);
  atomicAdd(o + 3, f.w * a);
}

// ---------------- layer-1 epilogue: h1 = elu(rst + b1) in place ------------
__global__ void k_elu_bias(float* __restrict__ h, const float* __restrict__ b) {
  const int i = blockIdx.x * blockDim.x + threadIdx.x;
  if (i >= NN * (HD1 / 4)) return;
  float4 v = *(float4*)(h + (size_t)i * 4);
  const float4 bb = *(const float4*)(b + (i % (HD1 / 4)) * 4);
  v.x += bb.x; v.y += bb.y; v.z += bb.z; v.w += bb.w;
  v.x = (v.x > 0.f) ? v.x : expm1f(v.x);
  v.y = (v.y > 0.f) ? v.y : expm1f(v.y);
  v.z = (v.z > 0.f) ? v.z : expm1f(v.z);
  v.w = (v.w > 0.f) ? v.w : expm1f(v.w);
  *(float4*)(h + (size_t)i * 4) = v;
}

// ---------------- final: out = rst2 + res2 + b2 (mean over 1 head = id) ----
__global__ void k_final_out(const float* __restrict__ rst2,
                            const float* __restrict__ res2,
                            const float* __restrict__ b2,
                            float* __restrict__ out) {
  const int i = blockIdx.x * blockDim.x + threadIdx.x;
  if (i >= NN * (HD2 / 4)) return;
  const float4 a  = *(const float4*)(rst2 + (size_t)i * 4);
  const float4 r  = *(const float4*)(res2 + (size_t)i * 4);
  const float4 bb = *(const float4*)(b2 + (i % (HD2 / 4)) * 4);
  float4 v;
  v.x = a.x + r.x + bb.x; v.y = a.y + r.y + bb.y;
  v.z = a.z + r.z + bb.z; v.w = a.w + r.w + bb.w;
  *(float4*)(out + (size_t)i * 4) = v;
}

// ---------------------------------------------------------------------------
extern "C" void kernel_launch(void* const* d_in, const int* in_sizes, int n_in,
                              void* d_out, int out_size, void* d_ws,
                              size_t ws_size, hipStream_t stream) {
  (void)in_sizes; (void)n_in; (void)out_size; (void)ws_size;

  const float* feat  = (const float*)d_in[0];
  const int*   src   = (const int*)d_in[1];
  const int*   dst   = (const int*)d_in[2];
  const float* W1    = (const float*)d_in[3];
  const float* al1   = (const float*)d_in[4];
  const float* ar1   = (const float*)d_in[5];
  const float* b1    = (const float*)d_in[6];
  const float* W2    = (const float*)d_in[7];
  const float* al2   = (const float*)d_in[8];
  const float* ar2   = (const float*)d_in[9];
  const float* b2    = (const float*)d_in[10];
  const float* resW2 = (const float*)d_in[11];
  float* out = (float*)d_out;

  // Workspace layout (floats). Layer-2 buffers overlay ft1 (dead after the
  // layer-1 aggregate). Peak = 29.6M floats = 118.4 MB.
  float* ws = (float*)d_ws;
  float* ft1    = ws;                                   // 12.8M
  float* ft2    = ws;                                   // 3.2M  (overlay)
  float* res2   = ws + (size_t)3200000;                 // 3.2M  (overlay)
  float* rst2   = ws + (size_t)6400000;                 // 3.2M  (overlay)
  float* e2     = ws + (size_t)9600000;                 // 0.8M  (overlay)
  float* el2    = ws + (size_t)10400000;                // 50k
  float* er2    = el2 + NN;
  float* emax2  = er2 + NN;
  float* denom2 = emax2 + NN;
  float* h1     = ws + (size_t)12800000;                // 12.8M (rst1 -> h1)
  float* e1     = ws + (size_t)25600000;                // 3.2M
  float* el1    = ws + (size_t)28800000;                // 200k
  float* er1    = el1 + (size_t)NN * H1C;
  float* emax1  = er1 + (size_t)NN * H1C;
  float* denom1 = emax1 + (size_t)NN * H1C;

  const int TB = 256;

  // ---- layer 1 init (0xFF pattern loses to any atomicMaxF update) ----
  hipMemsetAsync(emax1, 0xFF, (size_t)NN * H1C * sizeof(float), stream);
  hipMemsetAsync(denom1, 0x00, (size_t)NN * H1C * sizeof(float), stream);
  hipMemsetAsync(h1, 0x00, (size_t)NN * HD1 * sizeof(float), stream);

  // ---- layer 1 ----
  k_wmma_gemm<<<dim3(HD1 / 64, NN / 16), 128, 0, stream>>>(feat, W1, ft1,
                                                           FIN, HD1);
  k_el_er<<<(NN * H1C + TB - 1) / TB, TB, 0, stream>>>(ft1, al1, ar1, el1,
                                                       er1, H1C);
  k_edge_logits_max<<<(NE * H1C + TB - 1) / TB, TB, 0, stream>>>(
      src, dst, el1, er1, e1, emax1, H1C);
  k_edge_exp_sum<<<(NE * H1C + TB - 1) / TB, TB, 0, stream>>>(dst, e1, emax1,
                                                              denom1, H1C);
  {
    const long long work = (long long)NE * (HD1 / 4);
    k_edge_aggregate<<<(unsigned)((work + TB - 1) / TB), TB, 0, stream>>>(
        src, dst, ft1, e1, denom1, h1, H1C, HD1);
  }
  k_elu_bias<<<(NN * (HD1 / 4) + TB - 1) / TB, TB, 0, stream>>>(h1, b1);

  // ---- layer 2 init (safe: ft1 region dead now) ----
  hipMemsetAsync(emax2, 0xFF, (size_t)NN * sizeof(float), stream);
  hipMemsetAsync(denom2, 0x00, (size_t)NN * sizeof(float), stream);
  hipMemsetAsync(rst2, 0x00, (size_t)NN * HD2 * sizeof(float), stream);

  // ---- layer 2 ----
  k_wmma_gemm_dual<<<dim3(1, NN / 16), 128, 0, stream>>>(h1, W2, resW2, ft2,
                                                         res2, HD1, HD2);
  k_el_er<<<(NN * H2C + TB - 1) / TB, TB, 0, stream>>>(ft2, al2, ar2, el2,
                                                       er2, H2C);
  k_edge_logits_max<<<(NE * H2C + TB - 1) / TB, TB, 0, stream>>>(
      src, dst, el2, er2, e2, emax2, H2C);
  k_edge_exp_sum<<<(NE * H2C + TB - 1) / TB, TB, 0, stream>>>(dst, e2, emax2,
                                                              denom2, H2C);
  {
    const long long work = (long long)NE * (HD2 / 4);
    k_edge_aggregate<<<(unsigned)((work + TB - 1) / TB), TB, 0, stream>>>(
        src, dst, ft2, e2, denom2, rst2, H2C, HD2);
  }
  k_final_out<<<(NN * (HD2 / 4) + TB - 1) / TB, TB, 0, stream>>>(rst2, res2,
                                                                 b2, out);
}
